// SelfAttention_3788161155433
// MI455X (gfx1250) — matrix operand order repaired and checked by reading, not compiled, not run on hardware
//
#include <hip/hip_runtime.h>
#include <hip/hip_bf16.h>

// ---------------------------------------------------------------------------
// CDNA5 (gfx1250) wave32 WMMA self-attention:
//   qkv = x @ w_in + b_in -> split heads -> flash attention -> @ w_out + b_out
// - all matrix math: v_wmma_f32_16x16x32_bf16 (bf16 operands, f32 accum)
// - all LDS staging: global_load_async_to_lds_b128 (ASYNCcnt), double-buffered
// - softmax row reductions: DPP16 (no LDS traffic)
// - operands pre-converted/pre-transposed once so hot loops are pure copies
// ---------------------------------------------------------------------------

typedef __bf16 bf16;
typedef __attribute__((ext_vector_type(16))) __bf16 v16bf;
typedef __attribute__((ext_vector_type(8)))  __bf16 v8bf;
typedef __attribute__((ext_vector_type(8)))  float  v8f;

union V16U { v16bf v; v8bf h[2]; };

__device__ __forceinline__ v8f vzero8() {
  v8f z;
  #pragma unroll
  for (int i = 0; i < 8; ++i) z[i] = 0.0f;
  return z;
}

__device__ __forceinline__ v8f wmma_bf16(v16bf a, v16bf b, v8f c) {
  // (neg_a, A, neg_b, B, c_mod, C, reuse_a, reuse_b)
  return __builtin_amdgcn_wmma_f32_16x16x32_bf16(false, a, false, b, (short)0, c,
                                                 false, false);
}

// A-operand (16x32 bf16): lane holds row M=lane%16; 16 contiguous K values
// {2*klo..2*klo+15} with klo = (lane<16 ? 0 : 8).  base = LDS row pointer.
__device__ __forceinline__ v16bf load_a16(const bf16* base, int klo) {
  V16U u;
  u.h[0] = *(const v8bf*)(base + 2 * klo);
  u.h[1] = *(const v8bf*)(base + 2 * klo + 8);
  return u.v;
}

// B-operand (32x16 bf16), staged as B^T[n][k]: lane n reads its column's 16
// contiguous K values (lanes 0-15: K 0..15, lanes 16-31: K 16..31).
__device__ __forceinline__ v16bf load_b16(const bf16* p) {
  V16U u;
  u.h[0] = *(const v8bf*)(p);
  u.h[1] = *(const v8bf*)(p + 8);
  return u.v;
}

// ---- DPP16 reductions within 16-lane groups (rows of the C layout) --------
template <int CTRL>
__device__ __forceinline__ float dpp_mov(float x) {
  int xi = __builtin_bit_cast(int, x);
  int r = __builtin_amdgcn_update_dpp(0, xi, CTRL, 0xF, 0xF, false);
  return __builtin_bit_cast(float, r);
}
__device__ __forceinline__ float redmax16(float v) {
  v = fmaxf(v, dpp_mov<0xB1>(v));   // quad_perm(1,0,3,2)  == xor 1
  v = fmaxf(v, dpp_mov<0x4E>(v));   // quad_perm(2,3,0,1)  == xor 2
  v = fmaxf(v, dpp_mov<0x141>(v));  // row_half_mirror     (other quad of 8)
  v = fmaxf(v, dpp_mov<0x140>(v));  // row_mirror          (other half of 16)
  return v;
}
__device__ __forceinline__ float redsum16(float v) {
  v += dpp_mov<0xB1>(v);
  v += dpp_mov<0x4E>(v);
  v += dpp_mov<0x141>(v);
  v += dpp_mov<0x140>(v);
  return v;
}

// ---- async global -> LDS (16 bytes per lane), tracked by ASYNCcnt ---------
__device__ __forceinline__ void async_ld16(const bf16* lds_dst, const bf16* g) {
  unsigned int loff = (unsigned int)(unsigned long long)lds_dst;  // low 32b = LDS offset
  unsigned long long ga = (unsigned long long)g;
  asm volatile("global_load_async_to_lds_b128 %0, %1, off"
               :: "v"(loff), "v"(ga) : "memory");
}
__device__ __forceinline__ void wait_async0() {
  asm volatile("s_wait_asynccnt 0x0" ::: "memory");
}

// ---------------------------------------------------------------------------
// Prep kernels (run once, streaming): fp32 -> bf16, and weight transpose so
// GEMM B-operand staging is a contiguous async copy.
// ---------------------------------------------------------------------------
__global__ __launch_bounds__(256) void cvt_bf16_kernel(
    const float* __restrict__ src, bf16* __restrict__ dst) {
  size_t i = ((size_t)blockIdx.x * 256 + threadIdx.x) * 8;
  #pragma unroll
  for (int j = 0; j < 8; ++j) dst[i + j] = (bf16)src[i + j];
}

// wT[n][k] = (bf16) w[k][n]  (32x32 LDS tile transpose)
__global__ __launch_bounds__(256) void transpose_cvt_kernel(
    const float* __restrict__ w, bf16* __restrict__ wT, int K, int N) {
  __shared__ float t[32][33];
  const int k0 = blockIdx.x * 32, n0 = blockIdx.y * 32;
  const int tx = threadIdx.x & 31, ty = threadIdx.x >> 5;  // ty: 0..7
  #pragma unroll
  for (int r = ty; r < 32; r += 8)
    t[r][tx] = w[(size_t)(k0 + r) * N + n0 + tx];
  __syncthreads();
  #pragma unroll
  for (int r = ty; r < 32; r += 8)
    wT[(size_t)(n0 + r) * K + k0 + tx] = (bf16)t[tx][r];
}

// ---------------------------------------------------------------------------
// Tiled GEMM  C[M,N] = A[M,K] @ BT[N,K]^T + bias[N]   (A, BT already bf16)
// Block tile 128x64, BK=32; 8 waves, each wave: 16 rows x 64 cols (4 tiles).
// Async double-buffered LDS staging; one barrier per K step.
// QKV=true : write bf16 Q/K [B=4][H=16][S=2048][64], V transposed [B][H][64][S]
// QKV=false: write fp32 into outF [M][N]
// ---------------------------------------------------------------------------
template <bool QKV>
__global__ __launch_bounds__(256) void gemm_bf16_kernel(
    const bf16* __restrict__ A, const bf16* __restrict__ BT,
    const float* __restrict__ bias,
    bf16* __restrict__ qb, bf16* __restrict__ kb, bf16* __restrict__ vb,
    float* __restrict__ outF, int M, int N, int K) {
  __shared__ __align__(16) bf16 As[2][128 * 32];  // [row][k]
  __shared__ __align__(16) bf16 Bs[2][64 * 32];   // B^T: [n][k]

  const int tid  = threadIdx.x;
  const int wid  = tid >> 5;
  const int lane = tid & 31;
  const int ln   = lane & 15;
  const int half = lane >> 4;
  const int m0 = blockIdx.x * 128;
  const int n0 = blockIdx.y * 64;

  // staging coordinates: A tile = 512 16B-chunks (2/thread), B tile = 256 (1/thread)
  const int ar0 = (tid) >> 2,        ac0 = (tid & 3) * 8;         // rows 0..63
  const int ar1 = (256 + tid) >> 2,  ac1 = (tid & 3) * 8;         // rows 64..127
  const int br  = tid >> 2,          bc  = (tid & 3) * 8;         // n 0..63

  const int nk = K >> 5;

  // prologue: stage k-chunk 0 into buffer 0
  async_ld16(&As[0][ar0 * 32 + ac0], &A[(size_t)(m0 + ar0) * K + ac0]);
  async_ld16(&As[0][ar1 * 32 + ac1], &A[(size_t)(m0 + ar1) * K + ac1]);
  async_ld16(&Bs[0][br * 32 + bc],   &BT[(size_t)(n0 + br) * K + bc]);
  wait_async0();
  __syncthreads();

  v8f acc[4];
  #pragma unroll
  for (int t = 0; t < 4; ++t) acc[t] = vzero8();

  for (int kt = 0; kt < nk; ++kt) {
    const int cur = kt & 1, nxt = cur ^ 1;
    if (kt + 1 < nk) {
      const int k0 = (kt + 1) * 32;
      async_ld16(&As[nxt][ar0 * 32 + ac0], &A[(size_t)(m0 + ar0) * K + k0 + ac0]);
      async_ld16(&As[nxt][ar1 * 32 + ac1], &A[(size_t)(m0 + ar1) * K + k0 + ac1]);
      async_ld16(&Bs[nxt][br * 32 + bc],   &BT[(size_t)(n0 + br) * K + k0 + bc]);
    }

    v16bf a = load_a16(&As[cur][(wid * 16 + ln) * 32], half ? 8 : 0);
    #pragma unroll
    for (int ct = 0; ct < 4; ++ct) {
      v16bf b = load_b16(&Bs[cur][(ct * 16 + ln) * 32 + half * 16]);
      acc[ct] = wmma_bf16(a, b, acc[ct]);
    }

    if (kt + 1 < nk) {
      wait_async0();     // next buffer landed
      __syncthreads();   // everyone done reading cur
    }
  }

  // --- epilogue: C layout is VGPR i -> row i + 8*(lane/16), col = lane%16 ---
  #pragma unroll
  for (int ct = 0; ct < 4; ++ct) {
    int col = n0 + ct * 16 + ln;
    float bia = bias[col];
    #pragma unroll
    for (int i = 0; i < 8; ++i) {
      int grow = m0 + wid * 16 + i + 8 * half;
      float v = acc[ct][i] + bia;
      if (QKV) {
        int which = col >> 10;         // 0:Q 1:K 2:V
        int cc = col & 1023;
        int h  = cc >> 6;
        int d  = cc & 63;
        int bb = grow >> 11;           // /2048
        int ss = grow & 2047;
        bf16 bv = (bf16)v;
        if (which == 0)
          qb[((size_t)(bb * 16 + h) * 2048 + ss) * 64 + d] = bv;
        else if (which == 1)
          kb[((size_t)(bb * 16 + h) * 2048 + ss) * 64 + d] = bv;
        else  // V stored transposed: [B][H][64][S]
          vb[((size_t)(bb * 16 + h) * 64 + d) * 2048 + ss] = bv;
      } else {
        outF[(size_t)grow * N + col] = v;
      }
    }
  }
}

// ---------------------------------------------------------------------------
// Flash attention: grid (B*H=64, S/128=16), 256 threads = 8 waves.
// Wave w owns q-rows [q0 + 16w, q0 + 16w + 16); streams keys in chunks of 32
// with double-buffered async global->LDS staging.
// ---------------------------------------------------------------------------
__global__ __launch_bounds__(256) void attn_kernel(
    const bf16* __restrict__ Qb, const bf16* __restrict__ Kb,
    const bf16* __restrict__ Vt, bf16* __restrict__ Ob) {
  __shared__ __align__(16) bf16 Qs[128 * 64];
  __shared__ __align__(16) bf16 Ks[2][32 * 64];   // [kcol][d] (== B^T for QK^T)
  __shared__ __align__(16) bf16 Vts[2][64 * 32];  // V^T: [d][kcol]
  __shared__ __align__(16) bf16 Ps[8][16 * 32];   // per-wave P tile

  const int tid  = threadIdx.x;
  const int wid  = tid >> 5;
  const int lane = tid & 31;
  const int ln   = lane & 15;
  const int half = lane >> 4;
  const int bh = blockIdx.x;            // b*16 + h
  const int q0 = blockIdx.y * 128;
  const size_t rowbase = (size_t)bh * 2048;          // Q/K row base
  const size_t vtbase  = (size_t)bh * 64 * 2048;     // V^T base

  // per-thread staging coordinates (all chunks are 16B)
  const int krow = tid >> 3, kcol = (tid & 7) * 8;   // K: 32x64
  const int vrow = tid >> 2, vcol = (tid & 3) * 8;   // V^T: 64x32

  // --- stage Q tile 128x64 (async) ---
  #pragma unroll
  for (int c = 0; c < 4; ++c) {
    int cid  = c * 256 + tid;
    int row  = cid >> 3;               // 0..127
    int colc = (cid & 7) * 8;
    async_ld16(&Qs[row * 64 + colc], &Qb[(rowbase + q0 + row) * 64 + colc]);
  }
  // --- prefetch chunk 0 of K / V^T into buffer 0 ---
  async_ld16(&Ks[0][krow * 64 + kcol], &Kb[(rowbase + krow) * 64 + kcol]);
  async_ld16(&Vts[0][vrow * 32 + vcol], &Vt[vtbase + (size_t)vrow * 2048 + vcol]);
  wait_async0();
  __syncthreads();

  const int klo = half ? 8 : 0;
  v16bf aq0 = load_a16(&Qs[(wid * 16 + ln) * 64], klo);       // d 0..31
  v16bf aq1 = load_a16(&Qs[(wid * 16 + ln) * 64 + 32], klo);  // d 32..63

  float mrow[8], lrow[8];
  v8f acc[4];
  #pragma unroll
  for (int i = 0; i < 8; ++i) { mrow[i] = -__builtin_inff(); lrow[i] = 0.0f; }
  #pragma unroll
  for (int t = 0; t < 4; ++t) acc[t] = vzero8();

  for (int kc = 0; kc < 64; ++kc) {     // 64 chunks x 32 keys = S
    const int cur = kc & 1, nxt = cur ^ 1;
    // issue async stage of next chunk into the other buffer
    if (kc + 1 < 64) {
      const size_t kvn = rowbase + (size_t)(kc + 1) * 32;
      async_ld16(&Ks[nxt][krow * 64 + kcol], &Kb[(kvn + krow) * 64 + kcol]);
      async_ld16(&Vts[nxt][vrow * 32 + vcol],
                 &Vt[vtbase + (size_t)vrow * 2048 + (kc + 1) * 32 + vcol]);
    }

    // scores S(16x32) = Q(16x64) @ K^T, chained over two K=32 halves
    v8f s0 = vzero8(), s1 = vzero8();
    #pragma unroll
    for (int h2 = 0; h2 < 2; ++h2) {
      v16bf aq = h2 ? aq1 : aq0;
      v16bf bk0 = load_b16(&Ks[cur][ln * 64 + h2 * 32 + half * 16]);
      s0 = wmma_bf16(aq, bk0, s0);
      v16bf bk1 = load_b16(&Ks[cur][(16 + ln) * 64 + h2 * 32 + half * 16]);
      s1 = wmma_bf16(aq, bk1, s1);
    }

    // online softmax; C-layout row = i + 8*half is uniform per 16-lane group
    #pragma unroll
    for (int i = 0; i < 8; ++i) {
      float a0 = s0[i] * 0.125f;        // 1/sqrt(64)
      float a1 = s1[i] * 0.125f;
      float rm = fmaxf(redmax16(a0), redmax16(a1));
      float mn = fmaxf(mrow[i], rm);
      float sc = __expf(mrow[i] - mn);
      mrow[i] = mn;
      float p0 = __expf(a0 - mn);
      float p1 = __expf(a1 - mn);
      lrow[i] = lrow[i] * sc + redsum16(p0) + redsum16(p1);
      acc[0][i] *= sc; acc[1][i] *= sc; acc[2][i] *= sc; acc[3][i] *= sc;
      int prow = i + 8 * half;
      Ps[wid][prow * 32 + ln]      = (bf16)p0;
      Ps[wid][prow * 32 + 16 + ln] = (bf16)p1;
    }
    asm volatile("" ::: "memory");      // wave-private LDS: order store->load

    // O += P(16x32) @ V(32x64)
    v16bf ap = load_a16(&Ps[wid][ln * 32], klo);
    #pragma unroll
    for (int ct = 0; ct < 4; ++ct) {
      v16bf bv = load_b16(&Vts[cur][(ct * 16 + ln) * 32 + half * 16]);
      acc[ct] = wmma_bf16(ap, bv, acc[ct]);
    }

    if (kc + 1 < 64) {
      wait_async0();                    // next buffer landed
      __syncthreads();                  // everyone done reading cur
    }
  }

  // epilogue -> Ob [B][S][D=1024] with D = h*64 + d
  const int b = bh >> 4, h = bh & 15;
  #pragma unroll
  for (int i = 0; i < 8; ++i) {
    float inv = 1.0f / lrow[i];
    int srow = q0 + wid * 16 + i + 8 * half;
    size_t obase = ((size_t)(b * 2048 + srow)) * 1024 + h * 64;
    #pragma unroll
    for (int ct = 0; ct < 4; ++ct)
      Ob[obase + ct * 16 + ln] = (bf16)(acc[ct][i] * inv);
  }
}

// ---------------------------------------------------------------------------
extern "C" void kernel_launch(void* const* d_in, const int* in_sizes, int n_in,
                              void* d_out, int out_size, void* d_ws,
                              size_t ws_size, hipStream_t stream) {
  const float* x     = (const float*)d_in[0];  // [4,2048,1024]
  const float* w_in  = (const float*)d_in[1];  // [1024,3072]
  const float* b_in  = (const float*)d_in[2];  // [3072]
  const float* w_out = (const float*)d_in[3];  // [1024,1024]
  const float* b_out = (const float*)d_in[4];  // [1024]
  float* out = (float*)d_out;                  // [4,2048,1024]

  bf16* ws = (bf16*)d_ws;
  const size_t SZ = (size_t)4 * 16 * 2048 * 64;  // 8,388,608 elems per tensor
  bf16* Qb  = ws;
  bf16* Kb  = ws + SZ;
  bf16* Vt  = ws + 2 * SZ;          // transposed: [B][H][64][S]
  bf16* Ob  = ws + 3 * SZ;
  bf16* xb  = ws + 4 * SZ;          // x in bf16            [8192][1024]
  bf16* wiT = ws + 5 * SZ;          // w_in^T  bf16         [3072][1024]
  bf16* woT = wiT + (size_t)3072 * 1024;  // w_out^T bf16   [1024][1024]

  // one-shot operand prep (pure streaming)
  cvt_bf16_kernel<<<4096, 256, 0, stream>>>(x, xb);               // 8M elems
  transpose_cvt_kernel<<<dim3(32, 96), 256, 0, stream>>>(w_in, wiT, 1024, 3072);
  transpose_cvt_kernel<<<dim3(32, 32), 256, 0, stream>>>(w_out, woT, 1024, 1024);

  // qkv = x @ w_in + b_in  -> Q/K bf16 [B,H,S,64], V bf16 [B,H,64,S]
  gemm_bf16_kernel<true><<<dim3(64, 48), 256, 0, stream>>>(
      xb, wiT, b_in, Qb, Kb, Vt, nullptr, 8192, 3072, 1024);
  // flash attention -> Ob bf16 [B,S,1024]
  attn_kernel<<<dim3(64, 16), 256, 0, stream>>>(Qb, Kb, Vt, Ob);
  // out = Ob @ w_out + b_out (fp32)
  gemm_bf16_kernel<false><<<dim3(64, 16), 256, 0, stream>>>(
      Ob, woT, b_out, nullptr, nullptr, nullptr, out, 8192, 1024, 1024);
}
